// LoRALinear_74500502717012
// MI455X (gfx1250) — matrix operand order repaired and checked by reading, not compiled, not run on hardware
//
#include <hip/hip_runtime.h>

// ---------------------------------------------------------------------------
// LoRA Linear (fused) for MI455X / gfx1250, wave32 + v_wmma_f32_16x16x32_bf16
//   out = x @ W^T + b + 2 * ((x @ loraB^T) @ loraA^T)
// M=8192 (4x2048), N=4096, K=4096, r=16.
// Compute-bound (AI ~820 FLOP/B) -> bf16 WMMA inputs, f32 accumulation.
// ---------------------------------------------------------------------------

typedef __attribute__((ext_vector_type(16))) __bf16 v16bf;
typedef __attribute__((ext_vector_type(8)))  __bf16 v8bf;
typedef __attribute__((ext_vector_type(8)))  float  v8f;
typedef __attribute__((ext_vector_type(4)))  float  fvec4;

#define MT 8192
#define NT 4096
#define KT 4096
#define BM 128
#define BN 128
#define BK 32
#define LDH 40   // LDS row stride in bf16 elems: BK + 8 pad (80B, 16B aligned)

union V16u { v16bf v; v8bf h[2]; };

__device__ __forceinline__ v8bf cvt8(const float* f) {
  v8bf r;
#pragma unroll
  for (int i = 0; i < 8; ++i) r[i] = (__bf16)f[i];
  return r;
}

__global__ __launch_bounds__(256)
void lora_linear_wmma(const float* __restrict__ X, const float* __restrict__ W,
                      const float* __restrict__ Bias, const float* __restrict__ LA,
                      const float* __restrict__ LB, float* __restrict__ Out)
{
  __shared__ __align__(16) __bf16 sx [BM * LDH];  // x tile   (bf16)
  __shared__ __align__(16) __bf16 sw [BN * LDH];  // W tile   (bf16)
  __shared__ __align__(16) __bf16 slb[16 * LDH];  // loraB tile
  __shared__ __align__(16) __bf16 sxr[BM * LDH];  // xr (K=32, upper 16 zero)

  const int tid    = threadIdx.x;
  const int lane   = tid & 31;
  const int laneLo = lane & 15;
  const int laneHi = lane >> 4;
  const int wv     = tid >> 5;     // wave id 0..7
  const int wr     = wv & 3;       // wave row  (4)  -> 32 M rows each
  const int wc     = wv >> 2;      // wave col  (2)  -> 64 N cols each

  const int bm = blockIdx.y * BM;
  const int bn = blockIdx.x * BN;

  // ---- global tile loaders: 256 threads cover 128 rows x 32 cols (fp32) ----
  const int lr = tid >> 1;          // 0..127
  const int lc = (tid & 1) * 16;    // 0 or 16
  const float* xg = X + (size_t)(bm + lr) * KT + lc;
  const float* wg = W + (size_t)(bn + lr) * KT + lc;
  // loraB loader: threads 0..63 cover 16 rows x 32 cols
  const int lbr = tid >> 2;         // 0..15 (valid tid<64)
  const int lbc = (tid & 3) * 8;    // 0/8/16/24
  const float* lbg = LB + (size_t)lbr * KT + lbc;

  // zero sxr once: cols 16..31 stay zero (K padding for the rank-16 WMMA)
  for (int i = tid; i < BM * LDH; i += 256) sxr[i] = (__bf16)0.0f;

  union F16u { fvec4 v[4]; float f[16]; } rx, rw;
  union F8u  { fvec4 v[2]; float f[8];  } rlb;

  // prologue: stage K-tile 0 into registers
#pragma unroll
  for (int i = 0; i < 4; ++i) {
    rx.v[i] = *(const fvec4*)(xg + i * 4);
    rw.v[i] = *(const fvec4*)(wg + i * 4);
  }
  if (tid < 64) { rlb.v[0] = *(const fvec4*)(lbg); rlb.v[1] = *(const fvec4*)(lbg + 4); }

  v8f acc[2][4] = {};   // 2 M-frags x 4 N-frags per wave
  v8f xr[2]     = {};   // x @ loraB^T accumulators (waves with wc==0 only)

  const int ksteps = KT / BK;       // 128
  for (int kt = 0; kt < ksteps; ++kt) {
    // ---- convert staged fp32 -> bf16, store to LDS ----
    *(v8bf*)&sx[lr * LDH + lc]     = cvt8(rx.f);
    *(v8bf*)&sx[lr * LDH + lc + 8] = cvt8(rx.f + 8);
    *(v8bf*)&sw[lr * LDH + lc]     = cvt8(rw.f);
    *(v8bf*)&sw[lr * LDH + lc + 8] = cvt8(rw.f + 8);
    if (tid < 64) *(v8bf*)&slb[lbr * LDH + lbc] = cvt8(rlb.f);
    __syncthreads();

    // ---- software pipeline: issue next K-tile global loads now ----
    if (kt + 1 < ksteps) {
      xg += BK; wg += BK; lbg += BK;
#pragma unroll
      for (int i = 0; i < 4; ++i) {
        rx.v[i] = *(const fvec4*)(xg + i * 4);
        rw.v[i] = *(const fvec4*)(wg + i * 4);
      }
      if (tid < 64) { rlb.v[0] = *(const fvec4*)(lbg); rlb.v[1] = *(const fvec4*)(lbg + 4); }
    }

    // ---- gather WMMA fragments from LDS ----
    // A (16x32 bf16): lane m (0-15) -> row m, K 0..15; lanes 16-31 -> K 16..31
    V16u af[2];
#pragma unroll
    for (int am = 0; am < 2; ++am) {
      const int row = wr * 32 + am * 16 + laneLo;
      af[am].h[0] = *(const v8bf*)&sx[row * LDH + laneHi * 16];
      af[am].h[1] = *(const v8bf*)&sx[row * LDH + laneHi * 16 + 8];
    }
    // B (32x16 bf16): lane n holds column n; lanes 0-15 -> K 0..15, lanes 16-31 -> K 16..31
    V16u bfr[4];
#pragma unroll
    for (int bq = 0; bq < 4; ++bq) {
      const int col = wc * 64 + bq * 16 + laneLo;
      bfr[bq].h[0] = *(const v8bf*)&sw[col * LDH + laneHi * 16];
      bfr[bq].h[1] = *(const v8bf*)&sw[col * LDH + laneHi * 16 + 8];
    }

    // ---- main GEMM WMMAs: 8 per wave per K-step ----
#pragma unroll
    for (int am = 0; am < 2; ++am)
#pragma unroll
      for (int bq = 0; bq < 4; ++bq)
        acc[am][bq] = __builtin_amdgcn_wmma_f32_16x16x32_bf16(
            false, af[am].v, false, bfr[bq].v, (short)0, acc[am][bq], false, false);

    // ---- LoRA rank path: xr += x_tile @ loraB^T (wave-uniform branch) ----
    if (wc == 0) {
      V16u lbf;
      lbf.h[0] = *(const v8bf*)&slb[laneLo * LDH + laneHi * 16];
      lbf.h[1] = *(const v8bf*)&slb[laneLo * LDH + laneHi * 16 + 8];
#pragma unroll
      for (int am = 0; am < 2; ++am)
        xr[am] = __builtin_amdgcn_wmma_f32_16x16x32_bf16(
            false, af[am].v, false, lbf.v, (short)0, xr[am], false, false);
    }
    __syncthreads();
  }

  // ---- epilogue: acc += (2*xr) @ loraA^T  (K=16 padded to 32 with zeros) ----
  if (wc == 0) {
    // C/D layout: vgpr i -> row base + i + 8*laneHi, col(r) = laneLo
#pragma unroll
    for (int am = 0; am < 2; ++am) {
      const int mb = wr * 32 + am * 16 + laneHi * 8;
#pragma unroll
      for (int i = 0; i < 8; ++i)
        sxr[(mb + i) * LDH + laneLo] = (__bf16)(2.0f * xr[am][i]);
    }
  }
  __syncthreads();

  // loraA as B-matrix: column n = loraA[n, 0..15]; K 16..31 are zero -> lanes 16-31 all zero
  V16u laf[4];
#pragma unroll
  for (int bq = 0; bq < 4; ++bq) {
    V16u u; u.v = (v16bf)((__bf16)0.0f);
    if (laneHi == 0) {
      const float* p = LA + (size_t)(bn + wc * 64 + bq * 16 + laneLo) * 16;
#pragma unroll
      for (int j = 0; j < 16; ++j) u.v[j] = (__bf16)p[j];
    }
    laf[bq] = u;
  }
  V16u xf[2];
#pragma unroll
  for (int am = 0; am < 2; ++am) {
    const int row = wr * 32 + am * 16 + laneLo;
    xf[am].h[0] = *(const v8bf*)&sxr[row * LDH + laneHi * 16];
    xf[am].h[1] = *(const v8bf*)&sxr[row * LDH + laneHi * 16 + 8];
  }
#pragma unroll
  for (int am = 0; am < 2; ++am)
#pragma unroll
    for (int bq = 0; bq < 4; ++bq)
      acc[am][bq] = __builtin_amdgcn_wmma_f32_16x16x32_bf16(
          false, xf[am].v, false, laf[bq].v, (short)0, acc[am][bq], false, false);

  // ---- bias + store ----
#pragma unroll
  for (int bq = 0; bq < 4; ++bq) {
    const int n = bn + wc * 64 + bq * 16 + laneLo;
    const float bias = Bias[n];
#pragma unroll
    for (int am = 0; am < 2; ++am) {
      const int mb = bm + wr * 32 + am * 16 + laneHi * 8;
#pragma unroll
      for (int i = 0; i < 8; ++i)
        Out[(size_t)(mb + i) * NT + n] = acc[am][bq][i] + bias;
    }
  }
}

extern "C" void kernel_launch(void* const* d_in, const int* in_sizes, int n_in,
                              void* d_out, int out_size, void* d_ws, size_t ws_size,
                              hipStream_t stream) {
  (void)in_sizes; (void)n_in; (void)out_size; (void)d_ws; (void)ws_size;
  const float* x  = (const float*)d_in[0];   // [4,2048,4096]
  const float* W  = (const float*)d_in[1];   // [4096,4096]
  const float* b  = (const float*)d_in[2];   // [4096]
  const float* lA = (const float*)d_in[3];   // [4096,16]
  const float* lB = (const float*)d_in[4];   // [16,4096]
  float* out = (float*)d_out;                // [4,2048,4096]

  dim3 grid(NT / BN, MT / BM);               // 32 x 64 = 2048 blocks
  dim3 block(256);                           // 8 wave32 waves
  lora_linear_wmma<<<grid, block, 0, stream>>>(x, W, b, lA, lB, out);
}
